// MainModel_84799834292926
// MI455X (gfx1250) — compile-verified, matrix operand order checked by reading, not run on hardware
//
#include <hip/hip_runtime.h>
#include <hip/hip_bf16.h>
#include <math.h>

// ---------------------------------------------------------------------------
// Model dims (fixed by the reference)
#define BB   8
#define NN   1024
#define DD   256
#define NHH  4
#define HDD  64
#define KK   512

typedef __attribute__((ext_vector_type(16))) __bf16 v16bf;
typedef __attribute__((ext_vector_type(2)))  __bf16 bf16x2;
typedef __attribute__((ext_vector_type(8)))  float  v8f;

// CDNA5 16-bit A(16x32)/B(32x16) fragment K index for element e, lane l:
//   K = e + 8*(e>=8) + 8*(lane>=16)
__device__ __forceinline__ int kmap(int e, int lane) {
  return e + (e & 8) + ((lane & 16) >> 1);
}

__device__ __forceinline__ v8f wmma_bf16(v16bf a, v16bf b, v8f c) {
  return __builtin_amdgcn_wmma_f32_16x16x32_bf16(false, a, false, b, (short)0, c,
                                                 false, false);
}

// ---------------------------------------------------------------------------
// Generic GEMM: out[b,oc,n] = f( sum_ic W[oc,ic]*X[b,ic,n] + bias[oc] (+R) )
// X layout: [B, IC, N] fp32.  Block tile (MF*64)(OC) x 64(cols), 8 waves,
// K-step 32, double-buffered LDS, packed bf16x2 staging.
#define GF_RELU 1

template <int MF>   // M-fragments per wave (1 or 2); block M tile = MF*64
__global__ void __launch_bounds__(256)
gemm_kernel(const float* __restrict__ W, const float* __restrict__ X,
            const float* __restrict__ bias, const float* R, float* out,
            int OC, int IC, int flags, float* pool) {
  const int tid  = threadIdx.x;
  const int wave = tid >> 5, lane = tid & 31;
  const int ocBase  = blockIdx.x * (MF * 64);
  const int colBase = blockIdx.y * 64;      // 64 cols never straddle a batch
  const int b  = colBase >> 10;
  const int nb = colBase & 1023;

  __shared__ __bf16 As[2][MF * 64 * 32];    // [m][k]
  __shared__ __bf16 Bs[2][64 * 32];         // [n][k]

  const int mrow = (wave & 3) * (16 * MF);  // wave M offset in block tile
  const int ncol = (wave >> 2) * 32;        // wave N offset (2 frags)
  const int lrow = lane & 15;
  const int hi8  = (lane & 16) >> 1;        // +8 rows for high half

  auto stage = [&](int sb, int kb) {
#pragma unroll
    for (int j = 0; j < MF * 4; ++j) {      // A: MF*1024 bf16x2 pairs
      int p = tid + j * 256;
      int m = p >> 4, kp2 = (p & 15) * 2;
      float2 s = *(const float2*)(W + (size_t)(ocBase + m) * IC + kb + kp2);
      bf16x2 t; t[0] = (__bf16)s.x; t[1] = (__bf16)s.y;
      *(bf16x2*)&As[sb][m * 32 + kp2] = t;
    }
#pragma unroll
    for (int j = 0; j < 4; ++j) {           // B: 1024 pairs
      int p = tid + j * 256;
      int n = p & 63, kp2 = (p >> 6) * 2;
      const float* src = X + ((size_t)b * IC + kb + kp2) * NN + nb + n;
      bf16x2 t; t[0] = (__bf16)src[0]; t[1] = (__bf16)src[NN];
      *(bf16x2*)&Bs[sb][n * 32 + kp2] = t;
    }
  };

  v8f acc[MF][2];
#pragma unroll
  for (int mf = 0; mf < MF; ++mf) { acc[mf][0] = {}; acc[mf][1] = {}; }

  stage(0, 0);
  int buf = 0;
  for (int kb = 0; kb < IC; kb += 32) {
    __syncthreads();
    if (kb + 32 < IC) stage(buf ^ 1, kb + 32);

    v16bf a[MF], b0, b1;
    const __bf16* bp0 = &Bs[buf][(ncol + lrow) * 32];
    const __bf16* bp1 = &Bs[buf][(ncol + 16 + lrow) * 32];
#pragma unroll
    for (int e = 0; e < 16; ++e) {
      int k = kmap(e, lane);
      b0[e] = bp0[k]; b1[e] = bp1[k];
#pragma unroll
      for (int mf = 0; mf < MF; ++mf)
        a[mf][e] = As[buf][(mrow + mf * 16 + lrow) * 32 + k];
    }
#pragma unroll
    for (int mf = 0; mf < MF; ++mf) {
      acc[mf][0] = wmma_bf16(a[mf], b0, acc[mf][0]);
      acc[mf][1] = wmma_bf16(a[mf], b1, acc[mf][1]);
    }
    buf ^= 1;
  }

#pragma unroll
  for (int mf = 0; mf < MF; ++mf) {
#pragma unroll
    for (int f = 0; f < 2; ++f) {
      v8f av = acc[mf][f];
      int n = nb + ncol + f * 16 + lrow;
#pragma unroll
      for (int r = 0; r < 8; ++r) {
        int oc = ocBase + mrow + mf * 16 + r + hi8;
        float vv = av[r] + (bias ? bias[oc] : 0.0f);
        size_t oi = ((size_t)b * OC + oc) * NN + n;
        if (R) vv += R[oi];
        if (flags & GF_RELU) vv = fmaxf(vv, 0.0f);
        if (pool) atomicMax((int*)pool + (size_t)b * OC + oc, __float_as_int(vv));
        else out[oi] = vv;
      }
    }
  }
}

// ---------------------------------------------------------------------------
// Flash-style attention.  Q,K,V,O: [B, 256, N]; channel c = d*NH + h.
// grid (N/128, NH, B); 8 waves/block; wave owns 16 query rows; online softmax.
// Row max via batched shfl_xor butterflies (one wait per mask level);
// row sum via an extra "P x ones" WMMA accumulator (no shuffles at all).
__global__ void __launch_bounds__(256)
attn_kernel(const float* __restrict__ Q, const float* __restrict__ K,
            const float* __restrict__ V, float* __restrict__ O) {
  const int tid = threadIdx.x, wave = tid >> 5, lane = tid & 31;
  const int h = blockIdx.y, b = blockIdx.z;
  const int nBase = blockIdx.x * 128;
  const int lrow = lane & 15, hi8 = (lane & 16) >> 1;

  __shared__ __bf16 Qs[128 * 64];      // [n][d]
  __shared__ __bf16 Ks[2][32 * 64];    // [m][d]
  __shared__ __bf16 Vs[2][64 * 32];    // [d][m]
  __shared__ __bf16 Ps[8][16 * 32];    // per-wave P tile [n][m]

  const size_t cbase = (size_t)b * DD * NN;

#pragma unroll
  for (int j = 0; j < 16; ++j) {       // Qs: 4096 bf16x2 pairs (pack over d)
    int p = tid + j * 256;
    int n = p & 127, dp = p >> 7;      // dp in 0..31
    bf16x2 t;
    t[0] = (__bf16)Q[cbase + (size_t)((2 * dp)     * NHH + h) * NN + nBase + n];
    t[1] = (__bf16)Q[cbase + (size_t)((2 * dp + 1) * NHH + h) * NN + nBase + n];
    *(bf16x2*)&Qs[n * 64 + 2 * dp] = t;
  }
  __syncthreads();

  v16bf aQlo, aQhi, ones;
  {
    const __bf16* qp = Qs + (wave * 16 + lrow) * 64;
#pragma unroll
    for (int e = 0; e < 16; ++e) {
      int k = kmap(e, lane);
      aQlo[e] = qp[k]; aQhi[e] = qp[32 + k];
      ones[e] = (__bf16)1.0f;
    }
  }

  auto stageKV = [&](int sb, int mb) {
#pragma unroll
    for (int j = 0; j < 4; ++j) {      // Ks: 1024 pairs (pack over d)
      int p = tid + j * 256;
      int m = p & 31, dp = p >> 5;     // dp 0..31
      bf16x2 t;
      t[0] = (__bf16)K[cbase + (size_t)((2 * dp)     * NHH + h) * NN + mb + m];
      t[1] = (__bf16)K[cbase + (size_t)((2 * dp + 1) * NHH + h) * NN + mb + m];
      *(bf16x2*)&Ks[sb][m * 64 + 2 * dp] = t;
    }
#pragma unroll
    for (int j = 0; j < 4; ++j) {      // Vs: 1024 pairs (pack over m, float2)
      int p = tid + j * 256;
      int mp = p & 15, d = p >> 4;     // d 0..63
      float2 s = *(const float2*)&V[cbase + (size_t)(d * NHH + h) * NN + mb + 2 * mp];
      bf16x2 t; t[0] = (__bf16)s.x; t[1] = (__bf16)s.y;
      *(bf16x2*)&Vs[sb][d * 32 + 2 * mp] = t;
    }
  };

  v8f o0 = {}, o1 = {}, o2 = {}, o3 = {}, o4 = {};   // o4 = running row sums
  float rowmax[8];
#pragma unroll
  for (int r = 0; r < 8; ++r) rowmax[r] = -INFINITY;

  stageKV(0, 0);
  int buf = 0;
  for (int mb = 0; mb < NN; mb += 32) {
    __syncthreads();
    if (mb + 32 < NN) stageKV(buf ^ 1, mb + 32);

    v16bf b0lo, b0hi, b1lo, b1hi;
    {
      const __bf16* k0 = &Ks[buf][lrow * 64];
      const __bf16* k1 = &Ks[buf][(16 + lrow) * 64];
#pragma unroll
      for (int e = 0; e < 16; ++e) {
        int k = kmap(e, lane);
        b0lo[e] = k0[k];      b0hi[e] = k0[32 + k];
        b1lo[e] = k1[k];      b1hi[e] = k1[32 + k];
      }
    }
    v8f s0 = {}, s1 = {};
    s0 = wmma_bf16(aQlo, b0lo, s0);  s0 = wmma_bf16(aQhi, b0hi, s0);
    s1 = wmma_bf16(aQlo, b1lo, s1);  s1 = wmma_bf16(aQhi, b1hi, s1);

    // ---- tile row max: batched butterfly (8 independent DS ops per wait)
    float pm[8];
#pragma unroll
    for (int r = 0; r < 8; ++r) pm[r] = fmaxf(s0[r], s1[r]);
#pragma unroll
    for (int mask = 1; mask <= 8; mask <<= 1) {
      float t[8];
#pragma unroll
      for (int r = 0; r < 8; ++r) t[r] = __shfl_xor(pm[r], mask, 32);
#pragma unroll
      for (int r = 0; r < 8; ++r) pm[r] = fmaxf(pm[r], t[r]);
    }

    __bf16* pw = Ps[wave];
#pragma unroll
    for (int r = 0; r < 8; ++r) {
      float nm = fmaxf(rowmax[r], pm[r]);
      float sc = __expf(rowmax[r] - nm);
      rowmax[r] = nm;
      float p0 = __expf(s0[r] - nm);
      float p1 = __expf(s1[r] - nm);
      o0[r] *= sc; o1[r] *= sc; o2[r] *= sc; o3[r] *= sc; o4[r] *= sc;
      pw[(r + hi8) * 32 + lrow]      = (__bf16)p0;
      pw[(r + hi8) * 32 + 16 + lrow] = (__bf16)p1;
    }
    // re-read P in A-fragment layout (same-wave LDS, in-order DS ops)
    v16bf pa;
#pragma unroll
    for (int e = 0; e < 16; ++e) pa[e] = pw[lrow * 32 + kmap(e, lane)];
#pragma unroll
    for (int fj = 0; fj < 4; ++fj) {
      v16bf bv;
      const __bf16* vp = &Vs[buf][(fj * 16 + lrow) * 32];
#pragma unroll
      for (int e = 0; e < 16; ++e) bv[e] = vp[kmap(e, lane)];
      v8f& of = fj == 0 ? o0 : fj == 1 ? o1 : fj == 2 ? o2 : o3;
      of = wmma_bf16(pa, bv, of);
    }
    o4 = wmma_bf16(pa, ones, o4);      // row sums ride the matrix unit
    buf ^= 1;
  }

  float inv[8];
#pragma unroll
  for (int r = 0; r < 8; ++r) inv[r] = 1.0f / (o4[r] * 8.0f);  // /sqrt(64)

#pragma unroll
  for (int fj = 0; fj < 4; ++fj) {
    const v8f& of = fj == 0 ? o0 : fj == 1 ? o1 : fj == 2 ? o2 : o3;
    int d = fj * 16 + lrow;
#pragma unroll
    for (int r = 0; r < 8; ++r) {
      int n = nBase + wave * 16 + r + hi8;
      O[cbase + (size_t)(d * NHH + h) * NN + n] = of[r] * inv[r];
    }
  }
}

// ---------------------------------------------------------------------------
// Small VALU kernels
__global__ void kenc0_kernel(const float* kp, const float* sc,
                             const float* W, const float* bias, float* out) {
  int i = blockIdx.x * 256 + threadIdx.x;      // B*32*N
  if (i >= BB * 32 * NN) return;
  int n = i & 1023; int t = i >> 10; int oc = t & 31; int b = t >> 5;
  float x0 = (kp[((size_t)b * NN + n) * 2 + 0] - 320.0f) * (1.0f / 448.0f);
  float x1 = (kp[((size_t)b * NN + n) * 2 + 1] - 240.0f) * (1.0f / 448.0f);
  float x2 = sc[b * NN + n];
  float v = W[oc * 3 + 0] * x0 + W[oc * 3 + 1] * x1 + W[oc * 3 + 2] * x2 + bias[oc];
  out[i] = fmaxf(v, 0.0f);
}

__global__ void concat_kernel(const float* x, const float* msg, float* cat) {
  size_t i = (size_t)blockIdx.x * 256 + threadIdx.x;  // B*512*N
  if (i >= (size_t)BB * 512 * NN) return;
  int n = i & 1023; size_t t = i >> 10; int c = (int)(t & 511); int b = (int)(t >> 9);
  cat[i] = (c < 256) ? x[((size_t)b * 256 + c) * NN + n]
                     : msg[((size_t)b * 256 + (c - 256)) * NN + n];
}

__global__ void embT_kernel(const float* emb, float* embT) {
  int i = blockIdx.x * 256 + threadIdx.x;      // K*D
  if (i >= KK * DD) return;
  int d = i & 255, k = i >> 8;
  embT[k * DD + d] = emb[(size_t)d * KK + k];
}

__global__ void norms_kernel(const float* emb, float* norms) {
  int k = blockIdx.x * 256 + threadIdx.x;
  if (k >= KK) return;
  float s = 0.0f;
  for (int d = 0; d < DD; ++d) { float e = emb[(size_t)d * KK + k]; s += e * e; }
  norms[k] = s;
}

__global__ void argmin_kernel(const float* dots, const float* norms,
                              int* idx, float* outArg) {
  int bn = blockIdx.x; int b = bn >> 10, n = bn & 1023;
  __shared__ float sv[256]; __shared__ int si[256];
  float best = INFINITY; int bi = 0;
  for (int k = threadIdx.x; k < KK; k += 256) {
    float v = norms[k] - 2.0f * dots[((size_t)b * KK + k) * NN + n];
    if (v < best) { best = v; bi = k; }
  }
  sv[threadIdx.x] = best; si[threadIdx.x] = bi;
  __syncthreads();
  for (int s = 128; s > 0; s >>= 1) {
    if (threadIdx.x < s) {
      float ov = sv[threadIdx.x + s]; int oi = si[threadIdx.x + s];
      if (ov < sv[threadIdx.x] ||
          (ov == sv[threadIdx.x] && oi < si[threadIdx.x])) {
        sv[threadIdx.x] = ov; si[threadIdx.x] = oi;
      }
    }
    __syncthreads();
  }
  if (threadIdx.x == 0) { idx[bn] = si[0]; outArg[bn] = (float)si[0]; }
}

__global__ void zq_kernel(const float* emb, const int* idx, float* zq) {
  size_t i = (size_t)blockIdx.x * 256 + threadIdx.x;  // B*256*N
  if (i >= (size_t)BB * DD * NN) return;
  int n = i & 1023; size_t t = i >> 10; int d = (int)(t & 255); int b = (int)(t >> 8);
  zq[i] = emb[(size_t)d * KK + idx[b * NN + n]];
}

__global__ void zero_kernel(float* p, int n) {
  int i = blockIdx.x * 256 + threadIdx.x;
  if (i < n) p[i] = 0.0f;
}

__global__ void fc1_kernel(const float* pooled, const float* W,
                           const float* bias, float* out) {
  int o = blockIdx.x, b = blockIdx.y;     // grid (40, B)
  __shared__ float red[256];
  float s = 0.0f;
  for (int c = threadIdx.x; c < 2048; c += 256)
    s += pooled[b * 2048 + c] * W[(size_t)o * 2048 + c];
  red[threadIdx.x] = s; __syncthreads();
  for (int st = 128; st > 0; st >>= 1) {
    if (threadIdx.x < st) red[threadIdx.x] += red[threadIdx.x + st];
    __syncthreads();
  }
  if (threadIdx.x == 0) out[b * 40 + o] = fmaxf(red[0] + bias[o], 0.0f);
}

__global__ void pose_kernel(const float* f1, const float* Wr, const float* br,
                            const float* Wt, const float* bt, float* pose) {
  int t = blockIdx.x * 64 + threadIdx.x;
  if (t >= BB * 6) return;
  int b = t / 6, j = t % 6;
  const float* W; const float* bbp; int jj;
  if (j < 3) { W = Wt; bbp = bt; jj = j; }       // pose = [out_t, out_r]
  else       { W = Wr; bbp = br; jj = j - 3; }
  float s = bbp[jj];
  for (int c = 0; c < 40; ++c) s += f1[b * 40 + c] * W[jj * 40 + c];
  pose[b * 6 + j] = s;
}

// ---------------------------------------------------------------------------
extern "C" void kernel_launch(void* const* d_in, const int* in_sizes, int n_in,
                              void* d_out, int out_size, void* d_ws, size_t ws_size,
                              hipStream_t stream) {
  (void)n_in; (void)out_size; (void)ws_size;
  const float* descriptors = (const float*)d_in[0];
  const float* keypoints   = (const float*)d_in[1];
  const float* scores      = (const float*)d_in[2];
  // d_in[3] = image (shape only)
  const float *kW[4], *kb[4];
  for (int i = 0; i < 4; ++i) {
    kW[i] = (const float*)d_in[4 + 2 * i];
    kb[i] = (const float*)d_in[5 + 2 * i];
  }
  const int g0 = 12;
  // dict flatten order: insertion (Wq,bq,...) vs jax-sorted (Wk,Wm,Wq,Wv,bk,...)
  const bool sorted = (in_sizes[g0 + 1] == 65536);
  const float* conv1_W = (const float*)d_in[g0 + 9 * 12 + 0];
  const float* conv1_b = (const float*)d_in[g0 + 9 * 12 + 1];
  const float* fc1_W   = (const float*)d_in[g0 + 9 * 12 + 2];
  const float* fc1_b   = (const float*)d_in[g0 + 9 * 12 + 3];
  const float* fc3r_W  = (const float*)d_in[g0 + 9 * 12 + 4];
  const float* fc3r_b  = (const float*)d_in[g0 + 9 * 12 + 5];
  const float* fc3t_W  = (const float*)d_in[g0 + 9 * 12 + 6];
  const float* fc3t_b  = (const float*)d_in[g0 + 9 * 12 + 7];
  const float* emb     = (const float*)d_in[g0 + 9 * 12 + 8];

  // workspace layout (floats)
  float* wsf   = (float*)d_ws;
  float* x     = wsf;
  float* q     = wsf + 1 * 2097152;
  float* k     = wsf + 2 * 2097152;
  float* v     = wsf + 3 * 2097152;
  float* att   = wsf + 4 * 2097152;
  float* msg   = wsf + 5 * 2097152;
  float* cat   = wsf + 6 * 2097152;      // 4194304 (reused as dots)
  float* hid   = cat + 4194304;          // 4194304
  float* embT  = hid + 4194304;          // 131072
  float* norms = embT + 131072;          // 512
  int*   idxb  = (int*)(norms + 512);    // 8192
  float* pooled = norms + 512 + 8192;    // 16384
  float* f1out  = pooled + 16384;        // 320
  float* dots   = cat;

  float* out_pose = (float*)d_out;
  float* out_x    = out_pose + 48;
  float* out_zq   = out_x + 2097152;
  float* out_arg  = out_zq + 2097152;

  const dim3 blk(256);
  auto gemm = [&](const float* W, const float* bias, const float* X,
                  const float* R, float* out, int OC, int IC, int flags,
                  float* pool) {
    if (OC % 128 == 0) {
      dim3 grid(OC / 128, (BB * NN) / 64);
      gemm_kernel<2><<<grid, blk, 0, stream>>>(W, X, bias, R, out, OC, IC, flags, pool);
    } else {
      dim3 grid(OC / 64, (BB * NN) / 64);
      gemm_kernel<1><<<grid, blk, 0, stream>>>(W, X, bias, R, out, OC, IC, flags, pool);
    }
  };

  // ---- keypoint encoder: [3->32->64->128->256], last adds descriptors
  kenc0_kernel<<<(BB * 32 * NN) / 256, blk, 0, stream>>>(keypoints, scores,
                                                         kW[0], kb[0], q);
  gemm(kW[1], kb[1], q, nullptr, k, 64, 32, GF_RELU, nullptr);
  gemm(kW[2], kb[2], k, nullptr, v, 128, 64, GF_RELU, nullptr);
  gemm(kW[3], kb[3], v, descriptors, x, 256, 128, 0, nullptr);

  // ---- 9 GNN layers
  for (int l = 0; l < 9; ++l) {
    const int base = g0 + l * 12;
    const float *Wq, *bq, *Wk, *bk, *Wv, *bv, *Wm, *bm;
    if (!sorted) {
      Wq = (const float*)d_in[base + 0];  bq = (const float*)d_in[base + 1];
      Wk = (const float*)d_in[base + 2];  bk = (const float*)d_in[base + 3];
      Wv = (const float*)d_in[base + 4];  bv = (const float*)d_in[base + 5];
      Wm = (const float*)d_in[base + 6];  bm = (const float*)d_in[base + 7];
    } else {
      Wk = (const float*)d_in[base + 0];  Wm = (const float*)d_in[base + 1];
      Wq = (const float*)d_in[base + 2];  Wv = (const float*)d_in[base + 3];
      bk = (const float*)d_in[base + 4];  bm = (const float*)d_in[base + 5];
      bq = (const float*)d_in[base + 6];  bv = (const float*)d_in[base + 7];
    }
    const float* W0 = (const float*)d_in[base + 8];
    const float* b0 = (const float*)d_in[base + 9];
    const float* W1 = (const float*)d_in[base + 10];
    const float* b1 = (const float*)d_in[base + 11];

    gemm(Wq, bq, x, nullptr, q, 256, 256, 0, nullptr);
    gemm(Wk, bk, x, nullptr, k, 256, 256, 0, nullptr);
    gemm(Wv, bv, x, nullptr, v, 256, 256, 0, nullptr);
    attn_kernel<<<dim3(NN / 128, NHH, BB), blk, 0, stream>>>(q, k, v, att);
    gemm(Wm, bm, att, nullptr, msg, 256, 256, 0, nullptr);
    concat_kernel<<<(BB * 512 * NN) / 256, blk, 0, stream>>>(x, msg, cat);
    gemm(W0, b0, cat, nullptr, hid, 512, 512, GF_RELU, nullptr);
    gemm(W1, b1, hid, x, x, 256, 512, 0, nullptr);   // x += delta (in place)
  }

  // final x -> output
  hipMemcpyAsync(out_x, x, (size_t)2097152 * sizeof(float),
                 hipMemcpyDeviceToDevice, stream);

  // ---- nearest-embedding lookup
  embT_kernel<<<(KK * DD) / 256, blk, 0, stream>>>(emb, embT);
  gemm(embT, nullptr, x, nullptr, dots, KK, DD, 0, nullptr);
  norms_kernel<<<2, blk, 0, stream>>>(emb, norms);
  argmin_kernel<<<BB * NN, blk, 0, stream>>>(dots, norms, idxb, out_arg);
  zq_kernel<<<(BB * DD * NN) / 256, blk, 0, stream>>>(emb, idxb, out_zq);

  // ---- pose head: relu(conv1) -> maxpool fused in GEMM epilogue
  zero_kernel<<<64, blk, 0, stream>>>(pooled, BB * 2048);
  gemm(conv1_W, conv1_b, out_zq, nullptr, nullptr, 2048, 256, GF_RELU, pooled);
  fc1_kernel<<<dim3(40, BB), blk, 0, stream>>>(pooled, fc1_W, fc1_b, f1out);
  pose_kernel<<<1, dim3(64), 0, stream>>>(f1out, fc3r_W, fc3r_b,
                                          fc3t_W, fc3t_b, out_pose);
}
